// AttentionLayer_532575945339
// MI455X (gfx1250) — matrix operand order repaired and checked by reading, not compiled, not run on hardware
//
#include <hip/hip_runtime.h>
#include <hip/hip_bf16.h>
#include <math.h>

// ---------------------------------------------------------------------------
// AttentionLayer on gfx1250 (CDNA5): bf16 WMMA (f32 accumulate) throughout.
//   H=1024, NH=16, HD=64, B=2, S=2048  -> M = B*S = 4096 rows
// Round 4: vectorized phase-split GEMM staging (b128 loads, b64 LDS stores),
//          register-pipelined GEMM K-loop, double-buffered ASYNCcnt-pipelined
//          flash-attention tiles, wave-private P tile without block barrier.
// ---------------------------------------------------------------------------

typedef __bf16 bf16_t;
typedef __attribute__((ext_vector_type(16))) __bf16 v16bf;
typedef __attribute__((ext_vector_type(8)))  __bf16 v8bf;
typedef __attribute__((ext_vector_type(4)))  __bf16 v4bf;
typedef __attribute__((ext_vector_type(8)))  float  v8f;

#define H_DIM   1024
#define NHEADS  16
#define HEAD_D  64
#define BATCH   2
#define SEQ     2048
#define MTOT    (BATCH*SEQ)      // 4096
#define NKT     (SEQ/64)         // 32 key tiles
#define LN_EPS  1e-5f

// Async global->LDS staging (CDNA5 GLOBAL_LOAD_ASYNC_TO_LDS_B128, ASYNCcnt).
#if defined(__gfx1250__) && \
    __has_builtin(__builtin_amdgcn_global_load_async_to_lds_b128) && \
    __has_builtin(__builtin_amdgcn_s_wait_asynccnt)
#define USE_ASYNC_LDS 1
#else
#define USE_ASYNC_LDS 0
#endif

#if USE_ASYNC_LDS
typedef int v4i_async __attribute__((vector_size(16)));   // gcc vector kind = builtin prototype
static __device__ __forceinline__ void async_cp16(bf16_t* lds, const bf16_t* g) {
  __builtin_amdgcn_global_load_async_to_lds_b128(
      (__attribute__((address_space(1))) v4i_async*)
          (__attribute__((address_space(1))) void*)(void*)g,
      (__attribute__((address_space(3))) v4i_async*)
          (__attribute__((address_space(3))) void*)(void*)lds,
      /*offset=*/0, /*cpol=*/0);
}
#endif

// 16-element bf16 fragment from two 16-byte-aligned LDS chunks (ISA 7.12.2).
static __device__ __forceinline__ v16bf make_frag(const bf16_t* p0, const bf16_t* p1) {
  v8bf lo = *(const v8bf*)p0;
  v8bf hi = *(const v8bf*)p1;
  v16bf r;
#pragma unroll
  for (int i = 0; i < 8; ++i) { r[i] = lo[i]; r[i + 8] = hi[i]; }
  return r;
}

static __device__ __forceinline__ v8f wmma_bf16(v16bf a, v16bf b, v8f c) {
  return __builtin_amdgcn_wmma_f32_16x16x32_bf16(false, a, false, b, (short)0, c,
                                                 false, false);
}

static __device__ __forceinline__ v4bf cvt4(float4 v) {
  v4bf r; r[0] = (bf16_t)v.x; r[1] = (bf16_t)v.y; r[2] = (bf16_t)v.z; r[3] = (bf16_t)v.w;
  return r;
}

// ---------------------------------------------------------------------------
// Kernel 1: fused QKV projection.  grid = (M/64, H/64, 3), block = 128 (4 waves)
// Q,K head-split bf16 (b, nh, s, hd); V stored TRANSPOSED per head (b, nh, hd, s).
// Register-pipelined: tile k+1 float4 loads issue before the WMMA block.
// ---------------------------------------------------------------------------
__global__ __launch_bounds__(128)
void qkv_gemm_kernel(const float* __restrict__ x,
                     const float* __restrict__ Wq, const float* __restrict__ bq,
                     const float* __restrict__ Wk, const float* __restrict__ bk,
                     const float* __restrict__ Wv, const float* __restrict__ bv,
                     bf16_t* __restrict__ Qb, bf16_t* __restrict__ Kb,
                     bf16_t* __restrict__ Vb) {
  __shared__ alignas(16) bf16_t As[64][64];   // x tile, row-major (K contig)
  __shared__ alignas(16) bf16_t Bs[64][64];   // W tile transposed: [n][k]

  const int o = blockIdx.z;
  const float* Wm   = (o == 0) ? Wq : (o == 1) ? Wk : Wv;
  const float* bias = (o == 0) ? bq : (o == 1) ? bk : bv;
  bf16_t* outp      = (o == 0) ? Qb : (o == 1) ? Kb : Vb;

  const int t = threadIdx.x, wave = t >> 5, lane = t & 31;
  const int m0 = blockIdx.x * 64;
  const int n0 = blockIdx.y * 64;

  v8f acc[4];
#pragma unroll
  for (int n = 0; n < 4; ++n)
#pragma unroll
    for (int i = 0; i < 8; ++i) acc[n][i] = 0.f;

  const int arow = wave * 16 + (lane & 15);
  const int kh8  = (lane >> 4) * 8;
  const int kb16 = (lane >> 4) * 16;

  // per-thread staging coordinates: 8 float4 units per tile per array
  float4 ra[8], rb[8];
#pragma unroll
  for (int i = 0; i < 8; ++i) {               // prologue: load tile 0
    int unit = t + i * 128;
    int r = unit >> 4, c4 = (unit & 15) * 4;
    ra[i] = *(const float4*)&x [(size_t)(m0 + r) * H_DIM + c4];
    rb[i] = *(const float4*)&Wm[(size_t)r       * H_DIM + n0 + c4];
  }

  for (int k0 = 0; k0 < H_DIM; k0 += 64) {
    // ---- commit staged registers to LDS (b64 stores for A) ----
#pragma unroll
    for (int i = 0; i < 8; ++i) {
      int unit = t + i * 128;
      int r = unit >> 4, c4 = (unit & 15) * 4;
      *(v4bf*)&As[r][c4] = cvt4(ra[i]);
      v4bf w4 = cvt4(rb[i]);
      Bs[c4 + 0][r] = w4[0];                  // transpose: Bs[n][k]
      Bs[c4 + 1][r] = w4[1];
      Bs[c4 + 2][r] = w4[2];
      Bs[c4 + 3][r] = w4[3];
    }
    __syncthreads();

    // ---- issue next tile's global loads; they overlap the WMMAs below ----
    if (k0 + 64 < H_DIM) {
#pragma unroll
      for (int i = 0; i < 8; ++i) {
        int unit = t + i * 128;
        int r = unit >> 4, c4 = (unit & 15) * 4;
        ra[i] = *(const float4*)&x [(size_t)(m0 + r)       * H_DIM + k0 + 64 + c4];
        rb[i] = *(const float4*)&Wm[(size_t)(k0 + 64 + r)  * H_DIM + n0 + c4];
      }
      if (k0 + 128 < H_DIM) {                 // keep GL2 two tiles ahead
        __builtin_prefetch(&x [(size_t)(m0 + (t >> 1)) * H_DIM + k0 + 128 + (t & 1) * 32], 0, 1);
        __builtin_prefetch(&Wm[(size_t)(k0 + 128 + (t >> 1)) * H_DIM + n0 + (t & 1) * 32], 0, 1);
      }
    }

    // ---- 8 WMMAs on the staged tile ----
#pragma unroll
    for (int ks = 0; ks < 2; ++ks) {
      v16bf a = make_frag(&As[arow][ks * 32 + kh8], &As[arow][ks * 32 + 16 + kh8]);
#pragma unroll
      for (int n = 0; n < 4; ++n) {
        int bcol = n * 16 + (lane & 15);
        v16bf b = make_frag(&Bs[bcol][ks * 32 + kh8], &Bs[bcol][ks * 32 + 16 + kh8]);
        acc[n] = wmma_bf16(a, b, acc[n]);
      }
    }
    __syncthreads();
  }

  // store: C/D row = wave*16 + (lane>>4)*8 + v, col = lane&15
  const bool vmat = (o == 2);
  const int row_l = wave * 16 + (lane >> 4) * 8;
#pragma unroll
  for (int n = 0; n < 4; ++n) {
    int cg = n0 + n * 16 + (lane & 15);
    int hh = cg >> 6, dd = cg & 63;
    float bsc = bias[cg];
#pragma unroll
    for (int v = 0; v < 8; ++v) {
      int m = m0 + row_l + v;
      int bb = m >> 11;              // m / SEQ
      int ss = m & (SEQ - 1);
      size_t idx = vmat
          ? (((size_t)bb * NHEADS + hh) * HEAD_D + dd) * SEQ + ss   // (b,h,d,s)
          : (((size_t)bb * NHEADS + hh) * SEQ + ss) * HEAD_D + dd;  // (b,h,s,d)
      outp[idx] = (bf16_t)(acc[n][v] + bsc);
    }
  }
}

// ---------------------------------------------------------------------------
// Kernel 2: flash attention.  grid = (S/64, NH, B), block = 128 (4 waves).
// Double-buffered K/V tiles: ASYNCcnt DMA for tile it+1 overlaps compute of
// tile it.  One barrier + one async-wait per key tile.  P tile is wave-private
// (same-wave LDS ordering, no block barrier needed).
// ---------------------------------------------------------------------------
__global__ __launch_bounds__(128)
void attn_kernel(const bf16_t* __restrict__ Qb, const bf16_t* __restrict__ Kb,
                 const bf16_t* __restrict__ Vb, bf16_t* __restrict__ Cb) {
  __shared__ alignas(16) bf16_t Qs[64][64];          // [q][d]
  __shared__ alignas(16) bf16_t Ks[2][64][64];       // [key][d]
  __shared__ alignas(16) bf16_t Vs[2][64][64];       // [d][key] (V pre-transposed)
  __shared__ alignas(16) bf16_t Ps[4][16][64];       // per-wave P tile [q][key]

  const int t = threadIdx.x, wave = t >> 5, lane = t & 31;
  const int q0 = blockIdx.x * 64;
  const int h  = blockIdx.y;
  const int b  = blockIdx.z;
  const size_t head_base = ((size_t)b * NHEADS + h) * SEQ * HEAD_D;

  // ---- prologue: Q tile + key tile 0 into buffer 0 ----
#if USE_ASYNC_LDS
  {
    const bf16_t* gQ = Qb + head_base + (size_t)q0 * HEAD_D;   // flat 8KB
    const bf16_t* gK = Kb + head_base;                          // tile 0
    const bf16_t* gV = Vb + head_base;
#pragma unroll
    for (int i = 0; i < 4; ++i) {
      int u = i * 128 + t;                        // 512 units of 8 bf16 (16B)
      async_cp16(&Qs[0][0] + u * 8, gQ + u * 8);
      async_cp16(&Ks[0][0][0] + u * 8, gK + u * 8);
      int d = u >> 3, c = (u & 7) * 8;
      async_cp16(&Vs[0][d][c], gV + (size_t)d * SEQ + c);
    }
  }
#else
#pragma unroll
  for (int i = 0; i < 32; ++i) {
    int idx = t + i * 128;
    int r = idx >> 6, d = idx & 63;
    Qs[r][d]    = Qb[head_base + (size_t)(q0 + r) * HEAD_D + d];
    Ks[0][r][d] = Kb[head_base + (size_t)r * HEAD_D + d];
    Vs[0][r][d] = Vb[head_base + (size_t)r * SEQ + d];   // r=d-dim, d=s-dim here
  }
#endif

  float m_run[8], l_run[8];
  v8f ctx[4];
#pragma unroll
  for (int v = 0; v < 8; ++v) { m_run[v] = -1e30f; l_run[v] = 0.f; }
#pragma unroll
  for (int n = 0; n < 4; ++n)
#pragma unroll
    for (int i = 0; i < 8; ++i) ctx[n][i] = 0.f;

  const float scale = 0.125f;                     // 1/sqrt(64)
  const int arow = wave * 16 + (lane & 15);
  const int kh8  = (lane >> 4) * 8;
  const int kb16 = (lane >> 4) * 16;

  for (int it = 0; it < NKT; ++it) {
    const int cur = it & 1;
#if USE_ASYNC_LDS
    __builtin_amdgcn_s_wait_asynccnt(0);          // tile `it` DMA complete
#endif
    __syncthreads();                              // visible to all; old buffer free

    if (it + 1 < NKT) {                           // launch DMA for tile it+1
      const int kt2 = (it + 1) * 64;
      const int nxt = cur ^ 1;
#if USE_ASYNC_LDS
      const bf16_t* gK = Kb + head_base + (size_t)kt2 * HEAD_D;
      const bf16_t* gV = Vb + head_base + kt2;
#pragma unroll
      for (int i = 0; i < 4; ++i) {
        int u = i * 128 + t;
        async_cp16(&Ks[nxt][0][0] + u * 8, gK + u * 8);
        int d = u >> 3, c = (u & 7) * 8;
        async_cp16(&Vs[nxt][d][c], gV + (size_t)d * SEQ + c);
      }
#else
#pragma unroll
      for (int i = 0; i < 32; ++i) {
        int idx = t + i * 128;
        int r = idx >> 6, d = idx & 63;
        Ks[nxt][r][d] = Kb[head_base + (size_t)(kt2 + r) * HEAD_D + d];
        Vs[nxt][r][d] = Vb[head_base + (size_t)r * SEQ + kt2 + d];
      }
#endif
    }

    // ---- scores = Q * K^T (16x64 per wave, 2 WMMA k-steps) ----
    v8f sc[4];
#pragma unroll
    for (int n = 0; n < 4; ++n)
#pragma unroll
      for (int i = 0; i < 8; ++i) sc[n][i] = 0.f;
#pragma unroll
    for (int ks = 0; ks < 2; ++ks) {
      v16bf a = make_frag(&Qs[arow][ks * 32 + kh8], &Qs[arow][ks * 32 + 16 + kh8]);
#pragma unroll
      for (int n = 0; n < 4; ++n) {
        int kcol = n * 16 + (lane & 15);
        v16bf bf = make_frag(&Ks[cur][kcol][ks * 32 + kh8],
                             &Ks[cur][kcol][ks * 32 + 16 + kh8]);
        sc[n] = wmma_bf16(a, bf, sc[n]);
      }
    }

    // ---- online softmax: rows live at (lane>>4)*8 + v across half-wave ----
#pragma unroll
    for (int v = 0; v < 8; ++v) {
      float mx = -1e30f;
#pragma unroll
      for (int n = 0; n < 4; ++n) {
        float sv = sc[n][v] * scale;
        sv = fminf(fmaxf(sv, -10000.f), 10000.f); // reference clamp
        sc[n][v] = sv;
        mx = fmaxf(mx, sv);
      }
      for (int off = 1; off < 16; off <<= 1)      // reduce 16 lanes of half
        mx = fmaxf(mx, __shfl_xor(mx, off, 32));
      float mnew = fmaxf(m_run[v], mx);
      float alpha = __expf(m_run[v] - mnew);
      int prow = (lane >> 4) * 8 + v;
      float s = 0.f;
#pragma unroll
      for (int n = 0; n < 4; ++n) {
        float p = __expf(sc[n][v] - mnew);
        s += p;
        Ps[wave][prow][n * 16 + (lane & 15)] = (bf16_t)p;
      }
      for (int off = 1; off < 16; off <<= 1) s += __shfl_xor(s, off, 32);
      l_run[v] = l_run[v] * alpha + s;
      m_run[v] = mnew;
#pragma unroll
      for (int n = 0; n < 4; ++n) ctx[n][v] *= alpha;
    }
    // Ps is wave-private: same-wave LDS ordering suffices, no block barrier.

    // ---- ctx += P * V (2 WMMA k-steps over 64 keys) ----
#pragma unroll
    for (int ks = 0; ks < 2; ++ks) {
      v16bf a = make_frag(&Ps[wave][lane & 15][ks * 32 + kh8],
                          &Ps[wave][lane & 15][ks * 32 + 16 + kh8]);
#pragma unroll
      for (int n = 0; n < 4; ++n) {
        int dcol = n * 16 + (lane & 15);
        v16bf bf = make_frag(&Vs[cur][dcol][ks * 32 + kh8],
                             &Vs[cur][dcol][ks * 32 + 16 + kh8]);
        ctx[n] = wmma_bf16(a, bf, ctx[n]);
      }
    }
  }

  // epilogue: normalize by l, store merged-head layout (b, s, h*64+d)
#pragma unroll
  for (int v = 0; v < 8; ++v) {
    int row_l = wave * 16 + (lane >> 4) * 8 + v;
    int s = q0 + row_l;
    float inv = 1.f / l_run[v];
#pragma unroll
    for (int n = 0; n < 4; ++n) {
      int d = n * 16 + (lane & 15);
      Cb[((size_t)b * SEQ + s) * H_DIM + h * HEAD_D + d] = (bf16_t)(ctx[n][v] * inv);
    }
  }
}

// ---------------------------------------------------------------------------
// Kernel 3: output projection  out = ctx @ Wo + bo   (fp32 out to d_out)
// Same register-pipelined structure; A tile is already bf16 (v8bf copies).
// ---------------------------------------------------------------------------
__global__ __launch_bounds__(128)
void oproj_kernel(const bf16_t* __restrict__ Cb, const float* __restrict__ Wo,
                  const float* __restrict__ bo, float* __restrict__ out) {
  __shared__ alignas(16) bf16_t As[64][64];
  __shared__ alignas(16) bf16_t Bs[64][64];

  const int t = threadIdx.x, wave = t >> 5, lane = t & 31;
  const int m0 = blockIdx.x * 64;
  const int n0 = blockIdx.y * 64;

  v8f acc[4];
#pragma unroll
  for (int n = 0; n < 4; ++n)
#pragma unroll
    for (int i = 0; i < 8; ++i) acc[n][i] = 0.f;

  const int arow = wave * 16 + (lane & 15);
  const int kh8  = (lane >> 4) * 8;
  const int kb16 = (lane >> 4) * 16;

  v8bf   ca[4];                                 // A tile: 512 16B units / 128 thr
  float4 rb[8];                                 // B tile: fp32 -> bf16
#pragma unroll
  for (int i = 0; i < 4; ++i) {
    int u = t + i * 128;
    int r = u >> 3, c8 = (u & 7) * 8;
    ca[i] = *(const v8bf*)&Cb[(size_t)(m0 + r) * H_DIM + c8];
  }
#pragma unroll
  for (int i = 0; i < 8; ++i) {
    int unit = t + i * 128;
    int r = unit >> 4, c4 = (unit & 15) * 4;
    rb[i] = *(const float4*)&Wo[(size_t)r * H_DIM + n0 + c4];
  }

  for (int k0 = 0; k0 < H_DIM; k0 += 64) {
#pragma unroll
    for (int i = 0; i < 4; ++i) {
      int u = t + i * 128;
      int r = u >> 3, c8 = (u & 7) * 8;
      *(v8bf*)&As[r][c8] = ca[i];
    }
#pragma unroll
    for (int i = 0; i < 8; ++i) {
      int unit = t + i * 128;
      int r = unit >> 4, c4 = (unit & 15) * 4;
      v4bf w4 = cvt4(rb[i]);
      Bs[c4 + 0][r] = w4[0];
      Bs[c4 + 1][r] = w4[1];
      Bs[c4 + 2][r] = w4[2];
      Bs[c4 + 3][r] = w4[3];
    }
    __syncthreads();

    if (k0 + 64 < H_DIM) {                      // issue next tile's loads
#pragma unroll
      for (int i = 0; i < 4; ++i) {
        int u = t + i * 128;
        int r = u >> 3, c8 = (u & 7) * 8;
        ca[i] = *(const v8bf*)&Cb[(size_t)(m0 + r) * H_DIM + k0 + 64 + c8];
      }
#pragma unroll
      for (int i = 0; i < 8; ++i) {
        int unit = t + i * 128;
        int r = unit >> 4, c4 = (unit & 15) * 4;
        rb[i] = *(const float4*)&Wo[(size_t)(k0 + 64 + r) * H_DIM + n0 + c4];
      }
    }

#pragma unroll
    for (int ks = 0; ks < 2; ++ks) {
      v16bf a = make_frag(&As[arow][ks * 32 + kh8], &As[arow][ks * 32 + 16 + kh8]);
#pragma unroll
      for (int n = 0; n < 4; ++n) {
        int bcol = n * 16 + (lane & 15);
        v16bf b = make_frag(&Bs[bcol][ks * 32 + kh8], &Bs[bcol][ks * 32 + 16 + kh8]);
        acc[n] = wmma_bf16(a, b, acc[n]);
      }
    }
    __syncthreads();
  }

  const int row_l = wave * 16 + (lane >> 4) * 8;
#pragma unroll
  for (int n = 0; n < 4; ++n) {
    int cg = n0 + n * 16 + (lane & 15);
    float bsc = bo[cg];
#pragma unroll
    for (int v = 0; v < 8; ++v)
      out[(size_t)(m0 + row_l + v) * H_DIM + cg] = acc[n][v] + bsc;
  }
}

// ---------------------------------------------------------------------------
// Kernel 4: residual + LayerNorm, in-place on d_out.  grid = M rows, 256 thr.
// ---------------------------------------------------------------------------
__global__ __launch_bounds__(256)
void ln_kernel(const float* __restrict__ x, const float* __restrict__ gamma,
               const float* __restrict__ beta, float* __restrict__ y) {
  __shared__ float red[16];
  const int row = blockIdx.x;
  const int t = threadIdx.x, wave = t >> 5, lane = t & 31;

  float vals[4], s = 0.f, ss = 0.f;
#pragma unroll
  for (int i = 0; i < 4; ++i) {
    int c = t + i * 256;
    float v = y[(size_t)row * H_DIM + c] + x[(size_t)row * H_DIM + c];
    vals[i] = v; s += v; ss += v * v;
  }
  for (int off = 1; off < 32; off <<= 1) {
    s  += __shfl_xor(s,  off, 32);
    ss += __shfl_xor(ss, off, 32);
  }
  if (lane == 0) { red[wave] = s; red[8 + wave] = ss; }
  __syncthreads();
  if (t < 32) {
    float a  = (lane < 8) ? red[lane]     : 0.f;
    float b2 = (lane < 8) ? red[8 + lane] : 0.f;
    for (int off = 1; off < 8; off <<= 1) {
      a  += __shfl_xor(a,  off, 32);
      b2 += __shfl_xor(b2, off, 32);
    }
    if (lane == 0) { red[0] = a; red[1] = b2; }
  }
  __syncthreads();
  const float mean = red[0] * (1.f / H_DIM);
  const float var  = red[1] * (1.f / H_DIM) - mean * mean;
  const float rstd = rsqrtf(var + LN_EPS);
#pragma unroll
  for (int i = 0; i < 4; ++i) {
    int c = t + i * 256;
    y[(size_t)row * H_DIM + c] = (vals[i] - mean) * rstd * gamma[c] + beta[c];
  }
}

// ---------------------------------------------------------------------------
extern "C" void kernel_launch(void* const* d_in, const int* in_sizes, int n_in,
                              void* d_out, int out_size, void* d_ws, size_t ws_size,
                              hipStream_t stream) {
  (void)in_sizes; (void)n_in; (void)out_size; (void)ws_size;
  const float* x        = (const float*)d_in[0];
  const float* Wq       = (const float*)d_in[1];
  const float* bq       = (const float*)d_in[2];
  const float* Wk       = (const float*)d_in[3];
  const float* bk       = (const float*)d_in[4];
  const float* Wv       = (const float*)d_in[5];
  const float* bv       = (const float*)d_in[6];
  const float* Wo       = (const float*)d_in[7];
  const float* bo       = (const float*)d_in[8];
  const float* ln_gamma = (const float*)d_in[9];
  const float* ln_beta  = (const float*)d_in[10];
  float* out = (float*)d_out;

  const size_t NELEM = (size_t)MTOT * H_DIM;      // 4M elems
  bf16_t* Qb = (bf16_t*)d_ws;                     // 8 MB each, 32 MB total
  bf16_t* Kb = Qb + NELEM;
  bf16_t* Vb = Kb + NELEM;                        // V transposed: (b, nh, hd, s)
  bf16_t* Cb = Vb + NELEM;

  dim3 g1(MTOT / 64, H_DIM / 64, 3);
  qkv_gemm_kernel<<<g1, 128, 0, stream>>>(x, Wq, bq, Wk, bk, Wv, bv, Qb, Kb, Vb);

  dim3 g2(SEQ / 64, NHEADS, BATCH);
  attn_kernel<<<g2, 128, 0, stream>>>(Qb, Kb, Vb, Cb);

  dim3 g3(MTOT / 64, H_DIM / 64);
  oproj_kernel<<<g3, 128, 0, stream>>>(Cb, Wo, bo, out);

  ln_kernel<<<MTOT, 256, 0, stream>>>(x, ln_gamma, ln_beta, out);
}
